// AttentionLayer_75797582839949
// MI455X (gfx1250) — matrix operand order repaired and checked by reading, not compiled, not run on hardware
//
#include <hip/hip_runtime.h>
#include <stdint.h>

#define BSZ 256
#define NPG 128
#define HID 768
#define DG  512

typedef __attribute__((ext_vector_type(16))) __bf16 v16bf;
typedef __attribute__((ext_vector_type(8)))  float  v8f;

// gfx1250 async global->LDS path (ASYNCcnt), guarded: fall back to direct
// loads if this toolchain doesn't declare the builtins.
#if defined(__has_builtin)
#if __has_builtin(__builtin_amdgcn_global_load_async_to_lds_b128) && \
    __has_builtin(__builtin_amdgcn_s_wait_asynccnt)
#define USE_ASYNC_LDS 1
#endif
#endif
#ifndef USE_ASYNC_LDS
#define USE_ASYNC_LDS 0
#endif

#if USE_ASYNC_LDS
// The clang builtin's pointer params carry the HIP-unnameable __device__ /
// __shared__ LangAS, so declare the underlying LLVM intrinsic directly with
// target address-space-attributed pointers (same pattern composable_kernel
// uses for llvm.amdgcn.raw.buffer.load.lds). Pointee types are irrelevant
// under opaque pointers.
__device__ void
llvm_global_load_async_to_lds_b128(
    __attribute__((address_space(1))) const void* gaddr,
    __attribute__((address_space(3))) void* ldsaddr,
    int offset, int cpol) __asm("llvm.amdgcn.global.load.async.to.lds.b128");

__device__ __forceinline__ void async_copy16(const float* g, const float* lds_generic) {
  // inttoptr casts only: no (disallowed) generic->AS1/AS3 addrspace casts.
  __attribute__((address_space(1))) const void* gp =
      reinterpret_cast<__attribute__((address_space(1))) const void*>(
          (uintptr_t)g);
  __attribute__((address_space(3))) void* lp =
      reinterpret_cast<__attribute__((address_space(3))) void*>(
          (uint32_t)(uintptr_t)lds_generic);   // flat LDS addr low 32 bits = LDS offset
  llvm_global_load_async_to_lds_b128(gp, lp, 0, 0);
}
#endif

// ---------------------------------------------------------------------------
// C(MxN) = [relu]( A(MxK) @ B(KxN or NxK if BT) + bias ) [+ C if ACCUM]
// One 16x16 C tile per wave32; K stepped by 32 with v_wmma_f32_16x16x32_bf16.
// A/B fragments use contiguous k = kk + 16*half + s in both operands.
// ---------------------------------------------------------------------------
template<bool RELU, bool BT, bool ACCUM>
__global__ __launch_bounds__(128)
void gemm_bf16_wmma(const float* __restrict__ A, const float* __restrict__ Bm,
                    const float* __restrict__ bias, float* __restrict__ C,
                    int M, int N, int K) {
  const int lane = threadIdx.x & 31;
  const int wave = threadIdx.x >> 5;            // 4 waves per block
  const int ntn  = N >> 4;
  const int tile = blockIdx.x * 4 + wave;
  if (tile >= (M >> 4) * ntn) return;           // wave-uniform: EXEC stays all-1s
  const int mt   = tile / ntn;
  const int nt   = tile - mt * ntn;
  const int ml   = lane & 15;
  const int half = lane >> 4;                   // 0: lanes 0-15, 1: lanes 16-31

  const float* Arow = A + (size_t)(mt * 16 + ml) * K;
  const int n = nt * 16 + ml;

  v8f acc = {0.f, 0.f, 0.f, 0.f, 0.f, 0.f, 0.f, 0.f};

  for (int kk = 0; kk < K; kk += 32) {
    v16bf a, b;
    const int ka = kk + half * 16;
#pragma unroll
    for (int j = 0; j < 8; ++j) {
      a[j]     = (__bf16)Arow[ka + j];
      a[j + 8] = (__bf16)Arow[ka + 8 + j];
    }
    const int kb = kk + half * 16;
#pragma unroll
    for (int j = 0; j < 16; ++j) {
      const float bv = BT ? Bm[(size_t)n * K + (kb + j)]
                          : Bm[(size_t)(kb + j) * N + n];
      b[j] = (__bf16)bv;
    }
    acc = __builtin_amdgcn_wmma_f32_16x16x32_bf16(
        /*neg_a=*/false, a, /*neg_b=*/false, b,
        /*c_mod=*/(short)0, acc, /*reuse_a=*/false, /*reuse_b=*/false);
  }

  const float bcol = bias ? bias[n] : 0.f;
#pragma unroll
  for (int r = 0; r < 8; ++r) {
    const int row = mt * 16 + half * 8 + r;
    float v = acc[r] + bcol;
    if (RELU)  v = fmaxf(v, 0.f);
    if (ACCUM) v += C[(size_t)row * N + n];
    C[(size_t)row * N + n] = v;
  }
}

// ---------------------------------------------------------------------------
// Fused per-batch attention core (both heads a0/a1 at once):
//   e_i[k]  = x[b,k,:] . Qk_i[b,:]          (bias term is softmax-invariant)
//   attn_i  = softmax(e_i / sqrt(768))
//   xbar_i  = sum_k attn_i[k] * x[b,k,:]    (V-projection folded out after)
// One block of 256 threads (8 wave32s) per batch. Pass 1 double-buffers 8-key
// (24 KB) chunks of x into LDS with async global->LDS copies so the HBM stream
// overlaps the dot-product FMA chain; pass 2 re-reads x from L2 (100 MB << 192
// MB L2).
// ---------------------------------------------------------------------------
__global__ __launch_bounds__(256)
void attn_core(const float* __restrict__ x,
               const float* __restrict__ Qk0, const float* __restrict__ Qk1,
               float* __restrict__ xbar0, float* __restrict__ xbar1) {
  const int b    = blockIdx.x;
  const int t    = threadIdx.x;
  const int lane = t & 31;
  const int wv   = t >> 5;                       // 8 waves

  __shared__ float sq0[HID];
  __shared__ float sq1[HID];
  __shared__ float se0[NPG];
  __shared__ float se1[NPG];

  for (int c = t; c < HID; c += 256) {
    sq0[c] = Qk0[(size_t)b * HID + c];
    sq1[c] = Qk1[(size_t)b * HID + c];
  }
  __syncthreads();

  const float* xb = x + (size_t)b * NPG * HID;

#if USE_ASYNC_LDS
  // ---- Pass 1 (async staged): 16 chunks of 8 keys, double-buffered in LDS.
  __shared__ float sx[2][8 * HID];               // 2 x 24 KB
  // Stage one chunk: 8*768 floats contiguous; 256 threads x 6 x b128 each.
  auto stage = [&](int chunk, int buf) {
    const float* src = xb + (size_t)(chunk * 8) * HID;
#pragma unroll
    for (int i = 0; i < 6; ++i) {
      const int off = (t + i * 256) * 4;         // float index, 16B aligned
      async_copy16(src + off, &sx[buf][off]);
    }
  };
  stage(0, 0);
  for (int chunk = 0; chunk < NPG / 8; ++chunk) {
    const int buf = chunk & 1;
    if (chunk + 1 < NPG / 8) {
      stage(chunk + 1, buf ^ 1);                 // prefetch next chunk
      __builtin_amdgcn_s_wait_asynccnt(6);       // this wave's chunk landed
    } else {
      __builtin_amdgcn_s_wait_asynccnt(0);
    }
    __syncthreads();                             // all waves' stages landed
    const float* xr = &sx[buf][wv * HID];        // wave wv owns key chunk*8+wv
    float p0 = 0.f, p1 = 0.f;
#pragma unroll 4
    for (int c = lane; c < HID; c += 32) {
      const float xv = xr[c];
      p0 = fmaf(xv, sq0[c], p0);
      p1 = fmaf(xv, sq1[c], p1);
    }
#pragma unroll
    for (int off = 16; off > 0; off >>= 1) {
      p0 += __shfl_xor(p0, off, 32);
      p1 += __shfl_xor(p1, off, 32);
    }
    if (lane == 0) {
      const int k = chunk * 8 + wv;
      se0[k] = p0; se1[k] = p1;
    }
    __syncthreads();                             // done reading buf before reuse
  }
#else
  // ---- Pass 1 (fallback): direct global reads, wave wv owns keys wv, wv+8,...
  for (int k = wv; k < NPG; k += 8) {
    const float* xr = xb + (size_t)k * HID;
    if (k + 8 < NPG)
      __builtin_prefetch(xb + (size_t)(k + 8) * HID + lane * 4, 0, 0);
    float p0 = 0.f, p1 = 0.f;
#pragma unroll 4
    for (int c = lane; c < HID; c += 32) {
      const float xv = xr[c];
      p0 = fmaf(xv, sq0[c], p0);
      p1 = fmaf(xv, sq1[c], p1);
    }
#pragma unroll
    for (int off = 16; off > 0; off >>= 1) {
      p0 += __shfl_xor(p0, off, 32);
      p1 += __shfl_xor(p1, off, 32);
    }
    if (lane == 0) { se0[k] = p0; se1[k] = p1; }
  }
  __syncthreads();
#endif

  // Softmax over 128 keys (single wave), scale = 1/sqrt(HEAD_DIM)
  const float inv_scale = 0.03608439182435161f;  // 1/sqrt(768)
  if (wv == 0) {
    float m0 = -3.0e38f, m1 = -3.0e38f;
#pragma unroll
    for (int k = lane; k < NPG; k += 32) {
      m0 = fmaxf(m0, se0[k]);
      m1 = fmaxf(m1, se1[k]);
    }
#pragma unroll
    for (int off = 16; off > 0; off >>= 1) {
      m0 = fmaxf(m0, __shfl_xor(m0, off, 32));
      m1 = fmaxf(m1, __shfl_xor(m1, off, 32));
    }
    float s0 = 0.f, s1 = 0.f;
#pragma unroll
    for (int k = lane; k < NPG; k += 32) {
      const float w0 = __expf((se0[k] - m0) * inv_scale);
      const float w1 = __expf((se1[k] - m1) * inv_scale);
      se0[k] = w0; se1[k] = w1;
      s0 += w0; s1 += w1;
    }
#pragma unroll
    for (int off = 16; off > 0; off >>= 1) {
      s0 += __shfl_xor(s0, off, 32);
      s1 += __shfl_xor(s1, off, 32);
    }
    const float r0 = 1.f / s0, r1 = 1.f / s1;
#pragma unroll
    for (int k = lane; k < NPG; k += 32) { se0[k] *= r0; se1[k] *= r1; }
  }
  __syncthreads();

  // Pass 2: weighted mean of x rows (coalesced: thread t owns cols t, t+256, t+512)
  float a00 = 0.f, a01 = 0.f, a02 = 0.f;
  float a10 = 0.f, a11 = 0.f, a12 = 0.f;
  for (int k = 0; k < NPG; ++k) {
    const float w0 = se0[k], w1 = se1[k];
    const float x0 = xb[(size_t)k * HID + t];
    const float x1 = xb[(size_t)k * HID + t + 256];
    const float x2 = xb[(size_t)k * HID + t + 512];
    a00 = fmaf(w0, x0, a00); a01 = fmaf(w0, x1, a01); a02 = fmaf(w0, x2, a02);
    a10 = fmaf(w1, x0, a10); a11 = fmaf(w1, x1, a11); a12 = fmaf(w1, x2, a12);
  }
  xbar0[(size_t)b * HID + t]       = a00;
  xbar0[(size_t)b * HID + t + 256] = a01;
  xbar0[(size_t)b * HID + t + 512] = a02;
  xbar1[(size_t)b * HID + t]       = a10;
  xbar1[(size_t)b * HID + t + 256] = a11;
  xbar1[(size_t)b * HID + t + 512] = a12;
}

// ---------------------------------------------------------------------------
extern "C" void kernel_launch(void* const* d_in, const int* in_sizes, int n_in,
                              void* d_out, int out_size, void* d_ws, size_t ws_size,
                              hipStream_t stream) {
  const float* x      = (const float*)d_in[0];
  // d_in[1] = batch: repeat(arange(256),128) -> dense batch is a pure reshape.
  const float* gene   = (const float*)d_in[2];
  const float* bionic = (const float*)d_in[3];
  const float* fc0_w  = (const float*)d_in[4];
  const float* fc0_b  = (const float*)d_in[5];
  const float* fc1_w  = (const float*)d_in[6];
  const float* fc1_b  = (const float*)d_in[7];
  const float* a0_wq  = (const float*)d_in[8];
  const float* a0_bq  = (const float*)d_in[9];
  const float* a0_wk  = (const float*)d_in[10];
  const float* a0_wv  = (const float*)d_in[12];
  const float* a0_bv  = (const float*)d_in[13];
  const float* a0_wo  = (const float*)d_in[14];
  const float* a0_bo  = (const float*)d_in[15];
  const float* a1_wq  = (const float*)d_in[16];
  const float* a1_bq  = (const float*)d_in[17];
  const float* a1_wk  = (const float*)d_in[18];
  const float* a1_wv  = (const float*)d_in[20];
  const float* a1_bv  = (const float*)d_in[21];
  const float* a1_wo  = (const float*)d_in[22];
  const float* a1_bo  = (const float*)d_in[23];

  float* ws = (float*)d_ws;
  const size_t S = (size_t)BSZ * HID;            // 196608 floats per buffer
  float* g0  = ws;         float* g1  = g0 + S;
  float* Q0  = g1 + S;     float* Q1  = Q0 + S;
  float* Qk0 = Q1 + S;     float* Qk1 = Qk0 + S;
  float* xb0 = Qk1 + S;    float* xb1 = xb0 + S;
  float* t0  = xb1 + S;    float* t1  = t0 + S;
  float* out = (float*)d_out;

  const dim3 blk(128);
  const dim3 grd(((BSZ / 16) * (HID / 16) + 3) / 4);   // 192 blocks, 4 waves each

  // Input projections: g_i = relu(in @ fcW + fcB)
  gemm_bf16_wmma<true,  false, false><<<grd, blk, 0, stream>>>(gene,   fc0_w, fc0_b, g0, BSZ, HID, DG);
  gemm_bf16_wmma<true,  false, false><<<grd, blk, 0, stream>>>(bionic, fc1_w, fc1_b, g1, BSZ, HID, DG);
  // Q_i = g_i @ Wq + bq
  gemm_bf16_wmma<false, false, false><<<grd, blk, 0, stream>>>(g0, a0_wq, a0_bq, Q0, BSZ, HID, HID);
  gemm_bf16_wmma<false, false, false><<<grd, blk, 0, stream>>>(g1, a1_wq, a1_bq, Q1, BSZ, HID, HID);
  // Qk_i = Q_i @ Wk^T  (K-projection folded into the query)
  gemm_bf16_wmma<false, true,  false><<<grd, blk, 0, stream>>>(Q0, a0_wk, nullptr, Qk0, BSZ, HID, HID);
  gemm_bf16_wmma<false, true,  false><<<grd, blk, 0, stream>>>(Q1, a1_wk, nullptr, Qk1, BSZ, HID, HID);
  // Fused attention core: single streaming pass over x (the only big traffic)
  attn_core<<<dim3(BSZ), dim3(256), 0, stream>>>(x, Qk0, Qk1, xb0, xb1);
  // out_i = (xbar_i @ Wv + bv) @ Wo + bo ; final result accumulates both heads
  gemm_bf16_wmma<false, false, false><<<grd, blk, 0, stream>>>(xb0, a0_wv, a0_bv, t0, BSZ, HID, HID);
  gemm_bf16_wmma<false, false, false><<<grd, blk, 0, stream>>>(xb1, a1_wv, a1_bv, t1, BSZ, HID, HID);
  gemm_bf16_wmma<false, false, false><<<grd, blk, 0, stream>>>(t0, a0_wo, a0_bo, out, BSZ, HID, HID);
  gemm_bf16_wmma<false, false, true ><<<grd, blk, 0, stream>>>(t1, a1_wo, a1_bo, out, BSZ, HID, HID);
  (void)in_sizes; (void)n_in; (void)out_size; (void)ws_size;
}
